// PNAOutcomeModel_83794811945600
// MI455X (gfx1250) — matrix operand order repaired and checked by reading, not compiled, not run on hardware
//
#include <hip/hip_runtime.h>

// ---------------------------------------------------------------------------
// PNA GNN on MI455X (gfx1250): bf16 WMMA (16x16x32) for all GEMMs, fused
// edge-message GEMM + segment reductions via L2 atomics, fused post-MLP +
// LayerNorm + ELU per node tile. Weights pre-transposed to N-major bf16 so
// every WMMA fragment is built from contiguous 128-bit loads.
// ---------------------------------------------------------------------------

typedef __attribute__((ext_vector_type(16))) __bf16 v16bf;
typedef __attribute__((ext_vector_type(8)))  __bf16 v8bf;
typedef __attribute__((ext_vector_type(8)))  float  v8f;

#define DEVINL __device__ __forceinline__

DEVINL v8f v8f_zero() {
  v8f z;
#pragma unroll
  for (int i = 0; i < 8; ++i) z[i] = 0.0f;
  return z;
}

// Build a 16-element bf16 fragment from two contiguous 8-element (16B) chunks.
DEVINL v16bf load_frag16(const __bf16* p0, const __bf16* p1) {
  v8bf lo = *reinterpret_cast<const v8bf*>(p0);
  v8bf hi = *reinterpret_cast<const v8bf*>(p1);
  return __builtin_shufflevector(lo, hi, 0, 1, 2, 3, 4, 5, 6, 7,
                                 8, 9, 10, 11, 12, 13, 14, 15);
}

// float atomic max via sign-split integer atomics (accumulator init = -inf)
DEVINL void atomicMaxF(float* addr, float val) {
  if (val >= 0.0f) {
    atomicMax(reinterpret_cast<int*>(addr), __float_as_int(val));
  } else {
    atomicMin(reinterpret_cast<unsigned int*>(addr),
              static_cast<unsigned int>(__float_as_int(val)));
  }
}

// ---------------------------- utility kernels ------------------------------

__global__ void fill_f32_kernel(float* __restrict__ p, float v, int n) {
  int i = blockIdx.x * blockDim.x + threadIdx.x;
  if (i < n) p[i] = v;
}

// fp32 [K x N] row-major  ->  bf16 [N x K] (N-major, i.e. transposed)
__global__ void conv_transpose_kernel(const float* __restrict__ in,
                                      __bf16* __restrict__ out, int K, int N) {
  int i = blockIdx.x * blockDim.x + threadIdx.x;
  if (i < K * N) {
    int k = i / N, n = i % N;
    out[n * K + k] = (__bf16)in[i];
  }
}

__global__ void degree_kernel(const int* __restrict__ dst,
                              float* __restrict__ cnt, int E) {
  int e = blockIdx.x * blockDim.x + threadIdx.x;
  if (e < E) atomicAdd(&cnt[dst[e]], 1.0f);
}

__global__ void avglog_kernel(const float* __restrict__ cnt,
                              float* __restrict__ sum, int N) {
  int i = blockIdx.x * blockDim.x + threadIdx.x;
  float v = (i < N) ? logf(cnt[i] + 1.0f) : 0.0f;
#pragma unroll
  for (int off = 16; off > 0; off >>= 1) v += __shfl_down(v, off, 32);
  __shared__ float ws[8];
  int lane = threadIdx.x & 31, w = threadIdx.x >> 5;
  if (lane == 0) ws[w] = v;
  __syncthreads();
  if (threadIdx.x == 0) {
    float t = 0.0f;
    int nw = (int)(blockDim.x >> 5);
    for (int j = 0; j < nw; ++j) t += ws[j];
    atomicAdd(sum, t);
  }
}

__global__ void node_scalars_kernel(const float* __restrict__ cnt,
                                    const float* __restrict__ logsum,
                                    float* __restrict__ dc,
                                    float* __restrict__ amp,
                                    float* __restrict__ att, int N) {
  int i = blockIdx.x * blockDim.x + threadIdx.x;
  if (i < N) {
    float c = cnt[i];
    float d = fmaxf(c, 1.0f);
    float al = logsum[0] / (float)N;
    float l = logf(d + 1.0f);
    dc[i]  = d;
    amp[i] = l / al;
    att[i] = al / l;
  }
}

// AGG layout per node: [mean(0..F) | sum(F..2F) | max(2F..3F) | std(3F..4F)]
// During accumulation: sum -> slot1, max -> slot2 (init -inf), sum^2 -> slot3.
__global__ void init_agg_kernel(float* __restrict__ AGG, int N, int F) {
  long i = (long)blockIdx.x * blockDim.x + threadIdx.x;
  long total = (long)N * 4 * F;
  if (i < total) {
    int c = (int)(i % (4 * F));
    bool is_max = (c >= 2 * F) && (c < 3 * F);
    AGG[i] = is_max ? -__builtin_inff() : 0.0f;
  }
}

__global__ void agg_finalize_kernel(float* __restrict__ AGG,
                                    const float* __restrict__ cnt,
                                    const float* __restrict__ dc,
                                    int N, int F) {
  long i = (long)blockIdx.x * blockDim.x + threadIdx.x;
  long total = (long)N * F;
  if (i >= total) return;
  int n = (int)(i / F);
  int c = (int)(i % F);
  float* base = AGG + (long)n * 4 * F;
  float s   = base[F + c];
  float sq  = base[3 * F + c];
  float d   = dc[n];
  float mean = s / d;
  float msq  = sq / d;
  float var  = msq - mean * mean;
  base[c]         = mean;                                   // mean slot
  base[3 * F + c] = sqrtf(fmaxf(var, 0.0f) + 1e-5f);        // std slot
  float mx = base[2 * F + c];
  base[2 * F + c] = (cnt[n] > 0.0f) ? mx : 0.0f;            // max slot
}

// -------------------- fused edge GEMM + segment scatter --------------------
// Block = 128 threads = 4 waves, processes 64 edges (16 per wave).
// A = cat(h[dst], h[src]) [64 x 2F] staged bf16 in LDS; B = Wt bf16 [F x 2F]
// (N-major).  C (fp32) is scattered with atomics into AGG keyed by dst.
template <int F>
__global__ __launch_bounds__(128) void edge_msg_kernel(
    const float* __restrict__ h, const int* __restrict__ src,
    const int* __restrict__ dst, const __bf16* __restrict__ Wt,
    const float* __restrict__ bias, float* __restrict__ AGG, int E) {
  constexpr int KD  = 2 * F;    // reduction dim
  constexpr int NCT = F / 16;   // output column tiles
  __shared__ __bf16 sA[64 * KD];
  __shared__ int sDst[64];

  const int e0  = blockIdx.x * 64;
  const int tid = threadIdx.x;

  if (tid < 64) {
    int e = e0 + tid;
    sDst[tid] = (e < E) ? dst[e] : 0;
  }
  for (int idx = tid; idx < 64 * KD; idx += 128) {
    int r = idx / KD, c = idx % KD;
    int e = e0 + r;
    float v = 0.0f;
    if (e < E) {
      int node = (c < F) ? dst[e] : src[e];
      int cc   = (c < F) ? c : (c - F);
      v = h[(long)node * F + cc];
    }
    sA[idx] = (__bf16)v;
  }
  __syncthreads();

  const int wave = tid >> 5, lane = tid & 31;
  const int half = lane >> 4, m = lane & 15;
  const int row0 = wave * 16;  // this wave's 16 edges within the block

  v8f acc[NCT];
#pragma unroll
  for (int t = 0; t < NCT; ++t) acc[t] = v8f_zero();

  for (int k0 = 0; k0 < KD; k0 += 32) {
    // A fragment (ISA 16-bit A 16x32 layout): two contiguous 8-elem chunks.
    const __bf16* ar = &sA[(row0 + m) * KD + k0];
    v16bf a = load_frag16(ar + half * 8, ar + 16 + half * 8);
#pragma unroll
    for (int t = 0; t < NCT; ++t) {
      int n = t * 16 + m;  // output column
      // B fragment: two contiguous 8-elem chunks (same k order as A).
      const __bf16* br = &Wt[(long)n * KD + k0];
      v16bf b = load_frag16(br + half * 8, br + 16 + half * 8);
      acc[t] = __builtin_amdgcn_wmma_f32_16x16x32_bf16(
          false, a, false, b, (short)0, acc[t], false, false);
    }
  }

  // Scatter: C lane layout -> row = j + 8*half, col = lane%16
#pragma unroll
  for (int t = 0; t < NCT; ++t) {
#pragma unroll
    for (int j = 0; j < 8; ++j) {
      int er = row0 + j + 8 * half;
      int e  = e0 + er;
      if (e < E) {
        int d   = sDst[er];
        int col = t * 16 + m;
        float val = acc[t][j] + bias[col];
        float* base = AGG + (long)d * (4 * F);
        atomicAdd(base + F + col, val);            // sum
        atomicAdd(base + 3 * F + col, val * val);  // sum of squares
        atomicMaxF(base + 2 * F + col, val);       // max
      }
    }
  }
}

// ------------------- fused post MLP + lin + LN + ELU -----------------------
// Block = 128 threads = 4 waves, processes 16 nodes.
// A = [h | agg | amp*agg | att*agg] (16 x 13F) staged bf16 in LDS.
// postWt: bf16 [FO x 13F] (N-major), linWt: bf16 [FO x FO] (N-major).
template <int F, int FO>
__global__ __launch_bounds__(128) void post_node_kernel(
    const float* __restrict__ h, const float* __restrict__ AGG,
    const float* __restrict__ amp, const float* __restrict__ att,
    const __bf16* __restrict__ postWt, const float* __restrict__ postB,
    const __bf16* __restrict__ linWt, const float* __restrict__ linB,
    const float* __restrict__ lnG, const float* __restrict__ lnB,
    float* __restrict__ hout, int N) {
  constexpr int K13 = 13 * F;
  constexpr int NCT = FO / 16;
  __shared__ __bf16 sA[16 * K13];
  __shared__ float  sOut[16 * FO];
  __shared__ __bf16 sOutB[16 * FO];
  __shared__ float  sOut2[16 * FO];

  const int n0  = blockIdx.x * 16;
  const int tid = threadIdx.x;

  for (int idx = tid; idx < 16 * K13; idx += 128) {
    int r = idx / K13, c = idx % K13;
    int n = n0 + r;
    float v = 0.0f;
    if (n < N) {
      if (c < F) {
        v = h[(long)n * F + c];
      } else {
        int cc  = c - F;
        int blk = cc / (4 * F);  // 0: agg, 1: amp*agg, 2: att*agg
        int off = cc % (4 * F);
        float a = AGG[(long)n * 4 * F + off];
        v = (blk == 0) ? a : ((blk == 1) ? a * amp[n] : a * att[n]);
      }
    }
    sA[idx] = (__bf16)v;
  }
  __syncthreads();

  const int wave = tid >> 5, lane = tid & 31;
  const int half = lane >> 4, m = lane & 15;

  if (wave < NCT) {
    v8f acc = v8f_zero();
    const int col = wave * 16 + m;
    for (int k0 = 0; k0 < K13; k0 += 32) {
      const __bf16* ar = &sA[m * K13 + k0];
      v16bf a = load_frag16(ar + half * 8, ar + 16 + half * 8);
      const __bf16* br = &postWt[(long)col * K13 + k0];
      v16bf b = load_frag16(br + half * 8, br + 16 + half * 8);
      acc = __builtin_amdgcn_wmma_f32_16x16x32_bf16(
          false, a, false, b, (short)0, acc, false, false);
    }
#pragma unroll
    for (int j = 0; j < 8; ++j) {
      int r = j + 8 * half;
      sOut[r * FO + col] = acc[j] + postB[col];
    }
  }
  __syncthreads();

  for (int idx = tid; idx < 16 * FO; idx += 128) sOutB[idx] = (__bf16)sOut[idx];
  __syncthreads();

  if (wave < NCT) {
    v8f acc = v8f_zero();
    const int col = wave * 16 + m;
    for (int k0 = 0; k0 < FO; k0 += 32) {
      const __bf16* ar = &sOutB[m * FO + k0];
      v16bf a = load_frag16(ar + half * 8, ar + 16 + half * 8);
      const __bf16* br = &linWt[(long)col * FO + k0];
      v16bf b = load_frag16(br + half * 8, br + 16 + half * 8);
      acc = __builtin_amdgcn_wmma_f32_16x16x32_bf16(
          false, a, false, b, (short)0, acc, false, false);
    }
#pragma unroll
    for (int j = 0; j < 8; ++j) {
      int r = j + 8 * half;
      sOut2[r * FO + col] = acc[j] + linB[col];
    }
  }
  __syncthreads();

  // Row LayerNorm + ELU, one thread per node row
  if (tid < 16) {
    int n = n0 + tid;
    if (n < N) {
      float mu = 0.0f;
      for (int c = 0; c < FO; ++c) mu += sOut2[tid * FO + c];
      mu /= (float)FO;
      float var = 0.0f;
      for (int c = 0; c < FO; ++c) {
        float d = sOut2[tid * FO + c] - mu;
        var += d * d;
      }
      var /= (float)FO;
      float inv = rsqrtf(var + 1e-5f);
      for (int c = 0; c < FO; ++c) {
        float hv = (sOut2[tid * FO + c] - mu) * inv * lnG[c] + lnB[c];
        hout[(long)n * FO + c] = (hv > 0.0f) ? hv : (expf(hv) - 1.0f);
      }
    }
  }
}

// ------------------------- proj + LN + selected head -----------------------
__global__ void head_kernel(const float* __restrict__ h2,
                            const int* __restrict__ D,
                            const float* __restrict__ projW,
                            const float* __restrict__ projB,
                            const float* __restrict__ lnG,
                            const float* __restrict__ lnB,
                            const float* __restrict__ hW1, const float* __restrict__ hb1,
                            const float* __restrict__ hW2, const float* __restrict__ hb2,
                            const float* __restrict__ hWo, const float* __restrict__ hbo,
                            float* __restrict__ y, int N) {
  int n = blockIdx.x * blockDim.x + threadIdx.x;
  if (n >= N) return;
  float hr[32];
#pragma unroll
  for (int c = 0; c < 32; ++c) hr[c] = h2[(long)n * 32 + c];
  float phi[16];
#pragma unroll
  for (int j = 0; j < 16; ++j) {
    float s = projB[j];
    for (int c = 0; c < 32; ++c) s += hr[c] * projW[c * 16 + j];
    phi[j] = s;
  }
  float mu = 0.0f;
#pragma unroll
  for (int j = 0; j < 16; ++j) mu += phi[j];
  mu *= (1.0f / 16.0f);
  float var = 0.0f;
#pragma unroll
  for (int j = 0; j < 16; ++j) { float d = phi[j] - mu; var += d * d; }
  var *= (1.0f / 16.0f);
  float inv = rsqrtf(var + 1e-5f);
#pragma unroll
  for (int j = 0; j < 16; ++j) phi[j] = (phi[j] - mu) * inv * lnG[j] + lnB[j];

  int k = D[n];
  k = (k < 0) ? 0 : ((k > 2) ? 2 : k);
  float t1[32];
#pragma unroll
  for (int j = 0; j < 32; ++j) {
    float s = hb1[k * 32 + j];
    for (int c = 0; c < 16; ++c) s += phi[c] * hW1[(k * 16 + c) * 32 + j];
    t1[j] = fmaxf(s, 0.0f);
  }
  float t2[16];
#pragma unroll
  for (int j = 0; j < 16; ++j) {
    float s = hb2[k * 16 + j];
    for (int c = 0; c < 32; ++c) s += t1[c] * hW2[(k * 32 + c) * 16 + j];
    t2[j] = fmaxf(s, 0.0f);
  }
  float s = hbo[k];
#pragma unroll
  for (int c = 0; c < 16; ++c) s += t2[c] * hWo[k * 16 + c];
  y[n] = s;
}

// ------------------------------- launcher ----------------------------------

extern "C" void kernel_launch(void* const* d_in, const int* in_sizes, int n_in,
                              void* d_out, int out_size, void* d_ws, size_t ws_size,
                              hipStream_t stream) {
  (void)n_in; (void)out_size; (void)ws_size;
  const int F0 = 128, F1 = 64, F2 = 32;
  const int N = in_sizes[0] / F0;
  const int E = in_sizes[1] / 2;

  const float* x   = (const float*)d_in[0];
  const int*   ei  = (const int*)d_in[1];
  const int*   D   = (const int*)d_in[2];
  const float* preW0  = (const float*)d_in[3];
  const float* preB0  = (const float*)d_in[4];
  const float* postW0 = (const float*)d_in[5];
  const float* postB0 = (const float*)d_in[6];
  const float* linW0  = (const float*)d_in[7];
  const float* linB0  = (const float*)d_in[8];
  const float* lnG0   = (const float*)d_in[9];
  const float* lnB0   = (const float*)d_in[10];
  const float* preW1  = (const float*)d_in[11];
  const float* preB1  = (const float*)d_in[12];
  const float* postW1 = (const float*)d_in[13];
  const float* postB1 = (const float*)d_in[14];
  const float* linW1  = (const float*)d_in[15];
  const float* linB1  = (const float*)d_in[16];
  const float* lnG1   = (const float*)d_in[17];
  const float* lnB1   = (const float*)d_in[18];
  const float* projW  = (const float*)d_in[19];
  const float* projB  = (const float*)d_in[20];
  const float* plnG   = (const float*)d_in[21];
  const float* plnB   = (const float*)d_in[22];
  const float* hW1    = (const float*)d_in[23];
  const float* hb1    = (const float*)d_in[24];
  const float* hW2    = (const float*)d_in[25];
  const float* hb2    = (const float*)d_in[26];
  const float* hWo    = (const float*)d_in[27];
  const float* hbo    = (const float*)d_in[28];

  const int* src = ei;
  const int* dst = ei + E;

  // ---- workspace carve ----
  char* base = (char*)d_ws;
  size_t off = 0;
  auto carve = [&](size_t bytes) -> char* {
    char* p = base + off;
    off = (off + bytes + 255) & ~(size_t)255;
    return p;
  };
  float* cnt    = (float*)carve((size_t)N * 4);
  float* dc     = (float*)carve((size_t)N * 4);
  float* amp    = (float*)carve((size_t)N * 4);
  float* att    = (float*)carve((size_t)N * 4);
  float* logsum = (float*)carve(256);
  float* h1     = (float*)carve((size_t)N * F1 * 4);
  float* h2     = (float*)carve((size_t)N * F2 * 4);
  float* AGG    = (float*)carve((size_t)N * 4 * F0 * 4);  // reused for both layers
  __bf16* preW0t  = (__bf16*)carve((size_t)in_sizes[3] * 2);
  __bf16* postW0t = (__bf16*)carve((size_t)in_sizes[5] * 2);
  __bf16* linW0t  = (__bf16*)carve((size_t)in_sizes[7] * 2);
  __bf16* preW1t  = (__bf16*)carve((size_t)in_sizes[11] * 2);
  __bf16* postW1t = (__bf16*)carve((size_t)in_sizes[13] * 2);
  __bf16* linW1t  = (__bf16*)carve((size_t)in_sizes[15] * 2);

  const int T = 256;
  auto blocks = [&](long n) { return (unsigned)((n + T - 1) / T); };

  // ---- degree statistics ----
  fill_f32_kernel<<<blocks(N), T, 0, stream>>>(cnt, 0.0f, N);
  fill_f32_kernel<<<1, 32, 0, stream>>>(logsum, 0.0f, 1);
  degree_kernel<<<blocks(E), T, 0, stream>>>(dst, cnt, E);
  avglog_kernel<<<blocks(N), T, 0, stream>>>(cnt, logsum, N);
  node_scalars_kernel<<<blocks(N), T, 0, stream>>>(cnt, logsum, dc, amp, att, N);

  // ---- weight conversion fp32 -> bf16, transposed to N-major (tiny) ----
  conv_transpose_kernel<<<blocks(in_sizes[3]), T, 0, stream>>>(preW0, preW0t, 2 * F0, F0);
  conv_transpose_kernel<<<blocks(in_sizes[5]), T, 0, stream>>>(postW0, postW0t, 13 * F0, F1);
  conv_transpose_kernel<<<blocks(in_sizes[7]), T, 0, stream>>>(linW0, linW0t, F1, F1);
  conv_transpose_kernel<<<blocks(in_sizes[11]), T, 0, stream>>>(preW1, preW1t, 2 * F1, F1);
  conv_transpose_kernel<<<blocks(in_sizes[13]), T, 0, stream>>>(postW1, postW1t, 13 * F1, F2);
  conv_transpose_kernel<<<blocks(in_sizes[15]), T, 0, stream>>>(linW1, linW1t, F2, F2);

  const unsigned edgeBlocks = (unsigned)((E + 63) / 64);
  const unsigned nodeBlocks = (unsigned)((N + 15) / 16);

  // ---- layer 0: F=128 -> 64 ----
  init_agg_kernel<<<blocks((long)N * 4 * F0), T, 0, stream>>>(AGG, N, F0);
  edge_msg_kernel<128><<<edgeBlocks, 128, 0, stream>>>(x, src, dst, preW0t, preB0, AGG, E);
  agg_finalize_kernel<<<blocks((long)N * F0), T, 0, stream>>>(AGG, cnt, dc, N, F0);
  post_node_kernel<128, 64><<<nodeBlocks, 128, 0, stream>>>(
      x, AGG, amp, att, postW0t, postB0, linW0t, linB0, lnG0, lnB0, h1, N);

  // ---- layer 1: F=64 -> 32 ----
  init_agg_kernel<<<blocks((long)N * 4 * F1), T, 0, stream>>>(AGG, N, F1);
  edge_msg_kernel<64><<<edgeBlocks, 128, 0, stream>>>(h1, src, dst, preW1t, preB1, AGG, E);
  agg_finalize_kernel<<<blocks((long)N * F1), T, 0, stream>>>(AGG, cnt, dc, N, F1);
  post_node_kernel<64, 32><<<nodeBlocks, 128, 0, stream>>>(
      h1, AGG, amp, att, postW1t, postB1, linW1t, linB1, lnG1, lnB1, h2, N);

  // ---- projection + heads ----
  head_kernel<<<blocks(N), T, 0, stream>>>(h2, D, projW, projB, plnG, plnB,
                                           hW1, hb1, hW2, hb2, hWo, hbo,
                                           (float*)d_out, N);
}
